// Llama4DecoderLayer_25245817766059
// MI455X (gfx1250) — compile-verified, matrix operand order checked by reading, not run on hardware
//
#include <hip/hip_runtime.h>
#include <hip/hip_bf16.h>

#define DEVFN __device__ __forceinline__

typedef __attribute__((ext_vector_type(16))) __bf16 bf16x16;
typedef __attribute__((ext_vector_type(8)))  float  f32x8;

// ---- problem constants (match reference) ----
constexpr int B_   = 2;
constexpr int S_   = 1024;
constexpr int H_   = 2048;
constexpr int NH_  = 16;
constexpr int NKV_ = 4;
constexpr int D_   = 128;
constexpr int I_   = 1024;
constexpr int E_   = 16;
constexpr int TOK  = B_ * S_;                 // 2048 tokens
constexpr int QKVN = NH_ * D_ + 2 * NKV_ * D_; // 3072
constexpr int GQA  = NH_ / NKV_;              // 4
constexpr int MAXTILES = TOK / 16;            // 128 worst-case row tiles per expert
constexpr float SCALING_ = 0.08838834764831845f; // 128^-0.5
constexpr float EPS_ = 1e-6f;

// register blocking for the dense GEMM: each wave computes GM x GN 16x16 tiles
constexpr int GM = 4;   // 64 rows per wave
constexpr int GN = 2;   // 32 cols per wave; block of 4 waves covers 128 cols
constexpr int EN = 4;   // expert GEMM: 4 n-tiles per wave (64 cols); block covers 256 cols

// ---- helpers ----
DEVFN unsigned short f2bf(float x) {
  unsigned int u = __float_as_uint(x);
  u += 0x7fffu + ((u >> 16) & 1u);   // round to nearest even
  return (unsigned short)(u >> 16);
}

union BFrag {
  bf16x16 v;
  uint4 q[2];
};

// Fragment loader for 16x16x32 bf16 WMMA.
// A-matrix: base points at [row0][k0] of a row-major (ld) matrix; lane holds row=lane&15,
//   halves at k = kb..kb+7 and k = 16+kb..16+kb+7 with kb = 8*(lane>=16).
// B-matrix: identical pattern with "row" meaning the N (column) index of a [N][K] buffer.
DEVFN bf16x16 load_frag(const unsigned short* base, int ld) {
  int lane = threadIdx.x & 31;
  int r  = lane & 15;
  int kb = (lane >> 4) << 3;
  const unsigned short* p = base + (size_t)r * ld;
  BFrag f;
  f.q[0] = *(const uint4*)(p + kb);
  f.q[1] = *(const uint4*)(p + 16 + kb);
  return f.v;
}

DEVFN f32x8 wmma_bf16(bf16x16 a, bf16x16 b, f32x8 c) {
  return __builtin_amdgcn_wmma_f32_16x16x32_bf16(false, a, false, b, (short)0, c, false, false);
}

// ---- weight transpose + fp32 -> bf16 convert: in [K][N] row-major -> out [N][K] ----
__global__ __launch_bounds__(256) void transpose_convert_kernel(
    const float* __restrict__ in, unsigned short* __restrict__ out,
    int K, int N, size_t in_bstride, size_t out_bstride)
{
  __shared__ float tile[32][33];
  const float* ib = in + (size_t)blockIdx.z * in_bstride;
  unsigned short* ob = out + (size_t)blockIdx.z * out_bstride;
  int n0 = blockIdx.x * 32, k0 = blockIdx.y * 32;
  for (int i = threadIdx.y; i < 32; i += 8) {
    int k = k0 + i, n = n0 + threadIdx.x;
    tile[i][threadIdx.x] = (k < K && n < N) ? ib[(size_t)k * N + n] : 0.f;
  }
  __syncthreads();
  for (int i = threadIdx.y; i < 32; i += 8) {
    int n = n0 + i, k = k0 + threadIdx.x;
    if (n < N && k < K) ob[(size_t)n * K + k] = f2bf(tile[threadIdx.x][i]);
  }
}

// ---- RMSNorm: one block per row; bf16 out (+ optional fp32 copy) ----
__global__ __launch_bounds__(256) void rmsnorm_kernel(
    const float* __restrict__ x, const float* __restrict__ w,
    unsigned short* __restrict__ out_bf, float* __restrict__ out_f32)
{
  int row = blockIdx.x;
  const float* xr = x + (size_t)row * H_;
  float ss = 0.f;
  for (int j = threadIdx.x; j < H_; j += 256) { float v = xr[j]; ss += v * v; }
  __shared__ float red[256];
  red[threadIdx.x] = ss;
  __syncthreads();
  for (int s = 128; s > 0; s >>= 1) {
    if (threadIdx.x < s) red[threadIdx.x] += red[threadIdx.x + s];
    __syncthreads();
  }
  float inv = rsqrtf(red[0] / (float)H_ + EPS_);
  for (int j = threadIdx.x; j < H_; j += 256) {
    float v = xr[j] * inv * w[j];
    out_bf[(size_t)row * H_ + j] = f2bf(v);
    if (out_f32) out_f32[(size_t)row * H_ + j] = v;
  }
}

// ---- dense WMMA GEMM, register-blocked: C[M][N] = A[M][K] * Bt[N][K] (+ addend) ----
// grid: (N/128, M/64); block 128 (4 waves); wave w handles cols [ (bx*4+w)*32 .. +31 ],
// rows [ by*64 .. +63 ] as a 4x2 grid of 16x16 WMMA tiles sharing B fragments.
__global__ __launch_bounds__(128) void gemm_bf16_kernel(
    const unsigned short* __restrict__ A, const unsigned short* __restrict__ Bt,
    float* __restrict__ C, const float* __restrict__ addend,
    int M, int N, int K)
{
  int w = threadIdx.x >> 5;
  int mbase = blockIdx.y * (GM * 16);
  int nbase = (blockIdx.x * 4 + w) * (GN * 16);
  if (mbase >= M) return;
  f32x8 acc[GM][GN] = {};
  for (int k0 = 0; k0 < K; k0 += 32) {
    if (k0 + 256 < K) {
      __builtin_prefetch(Bt + (size_t)nbase * K + k0 + 256, 0, 1);       // global_prefetch_b8
      __builtin_prefetch(Bt + (size_t)(nbase + 16) * K + k0 + 256, 0, 1);
    }
    bf16x16 bf[GN];
    for (int j = 0; j < GN; ++j)
      bf[j] = load_frag(Bt + (size_t)(nbase + j * 16) * K + k0, K);
    for (int i = 0; i < GM; ++i) {
      bf16x16 af = load_frag(A + (size_t)(mbase + i * 16) * K + k0, K);
      for (int j = 0; j < GN; ++j)
        acc[i][j] = wmma_bf16(af, bf[j], acc[i][j]);
    }
  }
  int lane = threadIdx.x & 31;
  int c0   = lane & 15;
  int roff = (lane >> 4) << 3;
  for (int i = 0; i < GM; ++i) {
    for (int j = 0; j < GN; ++j) {
      int col = nbase + j * 16 + c0;
      for (int r = 0; r < 8; ++r) {
        int row = mbase + i * 16 + r + roff;
        float v = acc[i][j][r];
        if (addend) v += addend[(size_t)row * N + col];
        C[(size_t)row * N + col] = v;
      }
    }
  }
}

// ---- expert WMMA GEMM with row gather/scatter by token list, 1x4 n-blocking ----
// grid: (N/256, MAXTILES, E); block 128 (4 waves); wave w -> n-tiles (bx*16 + w*4) .. +3.
__global__ __launch_bounds__(128) void gemm_expert_kernel(
    const unsigned short* __restrict__ A, const unsigned short* __restrict__ BtAll,
    float* __restrict__ C, const int* __restrict__ tokenlist,
    const int* __restrict__ offsets, const int* __restrict__ counts,
    int N, int K)
{
  int e = blockIdx.z;
  int cnt = counts[e];
  int mtile = blockIdx.y;
  if (mtile * 16 >= cnt) return;   // uniform per block -> EXEC stays all-1 for WMMA
  int w = threadIdx.x >> 5;
  int nbase = (blockIdx.x * 16 + w * EN) * 16;
  const int* list = tokenlist + offsets[e];
  int lane = threadIdx.x & 31;
  int rowid = mtile * 16 + (lane & 15);
  int tok = list[(rowid < cnt) ? rowid : (cnt - 1)];
  const unsigned short* Ab = A + (size_t)tok * K;
  const unsigned short* Bb = BtAll + (size_t)e * N * K;
  int kb = (lane >> 4) << 3;
  f32x8 acc[EN] = {};
  for (int k0 = 0; k0 < K; k0 += 32) {
    if (k0 + 256 < K)
      __builtin_prefetch(Bb + (size_t)nbase * K + k0 + 256, 0, 1);
    BFrag fa;
    fa.q[0] = *(const uint4*)(Ab + k0 + kb);
    fa.q[1] = *(const uint4*)(Ab + k0 + 16 + kb);
    for (int j = 0; j < EN; ++j) {
      bf16x16 bf = load_frag(Bb + (size_t)(nbase + j * 16) * K + k0, K);
      acc[j] = wmma_bf16(fa.v, bf, acc[j]);
    }
  }
  int roff = (lane >> 4) << 3;
  for (int r = 0; r < 8; ++r) {
    int ridx = mtile * 16 + r + roff;
    if (ridx < cnt) {
      int trow = list[ridx];
      for (int j = 0; j < EN; ++j)
        C[(size_t)trow * N + nbase + j * 16 + (lane & 15)] = acc[j][r];
    }
  }
}

// ---- RoPE + per-head L2 norm; writes q [b][h][s][d] and k [b][kv][s][d] as bf16 ----
__global__ __launch_bounds__(32) void rope_l2norm_kernel(
    const float* __restrict__ qkv, const int* __restrict__ pos_ids,
    const float* __restrict__ freqs, unsigned short* __restrict__ qout,
    unsigned short* __restrict__ kout)
{
  int bid = blockIdx.x;
  int t  = bid / (NH_ + NKV_);
  int hh = bid % (NH_ + NKV_);
  int b = t / S_, s = t % S_;
  int pos = pos_ids[t];
  int lane = threadIdx.x;
  const float* src;
  unsigned short* dst;
  if (hh < NH_) {
    src = qkv + (size_t)t * QKVN + hh * D_;
    dst = qout + ((size_t)(b * NH_ + hh) * S_ + s) * D_;
  } else {
    int kv = hh - NH_;
    src = qkv + (size_t)t * QKVN + NH_ * D_ + kv * D_;
    dst = kout + ((size_t)(b * NKV_ + kv) * S_ + s) * D_;
  }
  float ra[2], rb[2];
  float ss = 0.f;
  for (int u = 0; u < 2; ++u) {
    int i = lane + u * 32;
    float a  = src[2 * i];
    float bb = src[2 * i + 1];
    float c  = freqs[((size_t)pos * (D_ / 2) + i) * 2 + 0];
    float sn = freqs[((size_t)pos * (D_ / 2) + i) * 2 + 1];
    ra[u] = a * c - bb * sn;
    rb[u] = a * sn + bb * c;
    ss += ra[u] * ra[u] + rb[u] * rb[u];
  }
  for (int off = 16; off > 0; off >>= 1) ss += __shfl_xor(ss, off, 32);
  float inv = rsqrtf(ss / (float)D_ + EPS_);
  for (int u = 0; u < 2; ++u) {
    int i = lane + u * 32;
    dst[2 * i]     = f2bf(ra[u] * inv);
    dst[2 * i + 1] = f2bf(rb[u] * inv);
  }
}

// ---- V transpose: qkv v-slice -> vT [b][kv][d][s] bf16 ----
__global__ __launch_bounds__(256) void v_transpose_kernel(
    const float* __restrict__ qkv, unsigned short* __restrict__ vT)
{
  int s  = blockIdx.x * 256 + threadIdx.x;
  int d  = blockIdx.y;
  int bk = blockIdx.z;                 // b*NKV+kv
  int b = bk / NKV_, kv = bk % NKV_;
  int t = b * S_ + s;
  float v = qkv[(size_t)t * QKVN + NH_ * D_ + NKV_ * D_ + kv * D_ + d];
  vT[((size_t)bk * D_ + d) * S_ + s] = f2bf(v);
}

// ---- WMMA flash attention: one wave per (b, h, 16-query tile), causal, GQA ----
__global__ __launch_bounds__(32) void attn_kernel(
    const unsigned short* __restrict__ qbf, const unsigned short* __restrict__ kbf,
    const unsigned short* __restrict__ vT, unsigned short* __restrict__ obf)
{
  int qtile = blockIdx.x;
  int h = blockIdx.y;
  int b = blockIdx.z;
  int kv = h / GQA;
  int qbase = qtile * 16;
  int lane = threadIdx.x;
  int col  = lane & 15;
  int roff = (lane >> 4) << 3;

  const unsigned short* qp = qbf + ((size_t)(b * NH_ + h) * S_ + qbase) * D_;
  const unsigned short* kp = kbf + ((size_t)(b * NKV_ + kv) * S_) * D_;
  const unsigned short* vp = vT  + ((size_t)(b * NKV_ + kv) * D_) * S_;

  bf16x16 qf[4];
  for (int kc = 0; kc < 4; ++kc) qf[kc] = load_frag(qp + kc * 32, D_);

  f32x8 o[8] = {};
  float m[8], l[8];
  for (int r = 0; r < 8; ++r) { m[r] = -3.0e38f; l[r] = 0.f; }

  __shared__ __align__(16) unsigned short Pld[16 * 32];

  int nch = (qbase + 47) >> 5;       // 32-key chunks covering keys 0..qbase+15
  for (int jc = 0; jc < nch; ++jc) {
    int key0 = jc * 32;
    f32x8 s0 = {}, s1 = {};
    for (int kc = 0; kc < 4; ++kc) {
      bf16x16 b0 = load_frag(kp + (size_t)key0 * D_ + kc * 32, D_);
      s0 = wmma_bf16(qf[kc], b0, s0);
      bf16x16 b1 = load_frag(kp + (size_t)(key0 + 16) * D_ + kc * 32, D_);
      s1 = wmma_bf16(qf[kc], b1, s1);
    }
    for (int r = 0; r < 8; ++r) {
      int rowg = qbase + r + roff;
      int c0 = key0 + col, c1 = key0 + 16 + col;
      s0[r] = (c0 > rowg) ? -3.0e38f : s0[r] * SCALING_;
      s1[r] = (c1 > rowg) ? -3.0e38f : s1[r] * SCALING_;
    }
    __syncthreads();
    for (int r = 0; r < 8; ++r) {
      float cm = fmaxf(s0[r], s1[r]);
      for (int off = 1; off < 16; off <<= 1) cm = fmaxf(cm, __shfl_xor(cm, off, 32));
      float mn = fmaxf(m[r], cm);
      float alpha = __expf(m[r] - mn);
      float p0 = __expf(s0[r] - mn);
      float p1 = __expf(s1[r] - mn);
      float rs = p0 + p1;
      for (int off = 1; off < 16; off <<= 1) rs += __shfl_xor(rs, off, 32);
      l[r] = l[r] * alpha + rs;
      m[r] = mn;
      for (int d = 0; d < 8; ++d) o[d][r] *= alpha;
      Pld[(r + roff) * 32 + col]      = f2bf(p0);
      Pld[(r + roff) * 32 + 16 + col] = f2bf(p1);
    }
    __syncthreads();
    bf16x16 pa = load_frag(Pld, 32);
    for (int d = 0; d < 8; ++d) {
      bf16x16 bv = load_frag(vp + (size_t)(d * 16) * S_ + key0, S_);
      o[d] = wmma_bf16(pa, bv, o[d]);
    }
  }
  float invl[8];
  for (int r = 0; r < 8; ++r) invl[r] = 1.0f / l[r];
  for (int d = 0; d < 8; ++d) {
    for (int r = 0; r < 8; ++r) {
      size_t tok = (size_t)b * S_ + qbase + r + roff;
      obf[tok * (NH_ * D_) + h * D_ + d * 16 + col] = f2bf(o[d][r] * invl[r]);
    }
  }
}

// ---- router logits: one block per token ----
__global__ __launch_bounds__(256) void router_logits_kernel(
    const float* __restrict__ x, const float* __restrict__ rw, float* __restrict__ logits)
{
  int t = blockIdx.x;
  float acc[E_];
  for (int e = 0; e < E_; ++e) acc[e] = 0.f;
  for (int j = threadIdx.x; j < H_; j += 256) {
    float v = x[(size_t)t * H_ + j];
    const float* r = rw + (size_t)j * E_;
    for (int e = 0; e < E_; ++e) acc[e] += v * r[e];
  }
  __shared__ float red[256 * E_];
  for (int e = 0; e < E_; ++e) red[threadIdx.x * E_ + e] = acc[e];
  __syncthreads();
  for (int s = 128; s > 0; s >>= 1) {
    if (threadIdx.x < s)
      for (int e = 0; e < E_; ++e) red[threadIdx.x * E_ + e] += red[(threadIdx.x + s) * E_ + e];
    __syncthreads();
  }
  if (threadIdx.x < E_) logits[(size_t)t * E_ + threadIdx.x] = red[threadIdx.x];
}

__global__ void zero16_kernel(int* counts) {
  if (threadIdx.x < E_) counts[threadIdx.x] = 0;
}

__global__ void top1_kernel(const float* __restrict__ logits, int* __restrict__ sidx,
                            float* __restrict__ sscore, int* __restrict__ counts)
{
  int t = blockIdx.x * blockDim.x + threadIdx.x;
  if (t >= TOK) return;
  float mv = logits[(size_t)t * E_];
  int mi = 0;
  for (int e = 1; e < E_; ++e) {
    float v = logits[(size_t)t * E_ + e];
    if (v > mv) { mv = v; mi = e; }
  }
  sidx[t] = mi;
  sscore[t] = 1.f / (1.f + __expf(-mv));
  atomicAdd(&counts[mi], 1);
}

__global__ void scan_kernel(const int* __restrict__ counts, int* __restrict__ offsets,
                            int* __restrict__ fillpos)
{
  if (threadIdx.x == 0) {
    int off = 0;
    for (int e = 0; e < E_; ++e) { offsets[e] = off; fillpos[e] = off; off += counts[e]; }
  }
}

__global__ void fill_kernel(const int* __restrict__ sidx, int* __restrict__ fillpos,
                            int* __restrict__ tokenlist)
{
  int t = blockIdx.x * blockDim.x + threadIdx.x;
  if (t >= TOK) return;
  int e = sidx[t];
  int p = atomicAdd(&fillpos[e], 1);
  tokenlist[p] = t;
}

__global__ void scale_tokens_kernel(const float* __restrict__ x, const float* __restrict__ sscore,
                                    unsigned short* __restrict__ xs)
{
  size_t i = (size_t)blockIdx.x * blockDim.x + threadIdx.x;
  if (i >= (size_t)TOK * H_) return;
  int t = (int)(i / H_);
  xs[i] = f2bf(x[i] * sscore[t]);
}

// gu [T][2I] -> act [T][I] = silu(gate)*up as bf16
__global__ void glu_act_kernel(const float* __restrict__ gu, unsigned short* __restrict__ act)
{
  size_t i = (size_t)blockIdx.x * blockDim.x + threadIdx.x;
  if (i >= (size_t)TOK * I_) return;
  size_t t = i / I_, j = i % I_;
  float g = gu[t * (size_t)(2 * I_) + j];
  float u = gu[t * (size_t)(2 * I_) + I_ + j];
  float s = g / (1.f + __expf(-g));
  act[i] = f2bf(s * u);
}

__global__ void final_add_kernel(const float* __restrict__ a, const float* __restrict__ b,
                                 const float* __restrict__ c, float* __restrict__ out)
{
  size_t i = (size_t)blockIdx.x * blockDim.x + threadIdx.x;
  if (i >= (size_t)TOK * H_) return;
  out[i] = a[i] + b[i] + c[i];
}

// ===================== host-side launch =====================
static inline size_t align256(size_t x) { return (x + 255) & ~(size_t)255; }

extern "C" void kernel_launch(void* const* d_in, const int* in_sizes, int n_in,
                              void* d_out, int out_size, void* d_ws, size_t ws_size,
                              hipStream_t stream) {
  (void)in_sizes; (void)n_in; (void)out_size; (void)ws_size;
  const float* hidden   = (const float*)d_in[0];
  const int*   pos_ids  = (const int*)d_in[1];
  const float* freqs    = (const float*)d_in[2];
  const float* ln1_w    = (const float*)d_in[3];
  const float* ln2_w    = (const float*)d_in[4];
  const float* wq       = (const float*)d_in[5];
  const float* wk       = (const float*)d_in[6];
  const float* wv       = (const float*)d_in[7];
  const float* wo       = (const float*)d_in[8];
  const float* router_w = (const float*)d_in[9];
  const float* gup      = (const float*)d_in[10];
  const float* dnp      = (const float*)d_in[11];
  const float* sh_gate  = (const float*)d_in[12];
  const float* sh_up    = (const float*)d_in[13];
  const float* sh_down  = (const float*)d_in[14];
  float* out = (float*)d_out;

  // workspace carve-out
  char* p = (char*)d_ws;
  auto take = [&](size_t bytes) { char* r = p; p += align256(bytes); return (void*)r; };
  unsigned short* wqkv_t  = (unsigned short*)take((size_t)QKVN * H_ * 2);
  unsigned short* wo_t    = (unsigned short*)take((size_t)H_ * H_ * 2);
  unsigned short* shgu_t  = (unsigned short*)take((size_t)(2 * I_) * H_ * 2);
  unsigned short* shdn_t  = (unsigned short*)take((size_t)H_ * I_ * 2);
  unsigned short* guexp_t = (unsigned short*)take((size_t)E_ * (2 * I_) * H_ * 2);
  unsigned short* dnexp_t = (unsigned short*)take((size_t)E_ * H_ * I_ * 2);
  unsigned short* h1_bf   = (unsigned short*)take((size_t)TOK * H_ * 2);
  float*          qkv     = (float*)take((size_t)TOK * QKVN * 4);
  unsigned short* q_bf    = (unsigned short*)take((size_t)B_ * NH_ * S_ * D_ * 2);
  unsigned short* k_bf    = (unsigned short*)take((size_t)B_ * NKV_ * S_ * D_ * 2);
  unsigned short* vT_bf   = (unsigned short*)take((size_t)B_ * NKV_ * D_ * S_ * 2);
  unsigned short* attn_bf = (unsigned short*)take((size_t)TOK * H_ * 2);
  float*          hid2    = (float*)take((size_t)TOK * H_ * 4);
  float*          h2      = (float*)take((size_t)TOK * H_ * 4);
  unsigned short* h2_bf   = (unsigned short*)take((size_t)TOK * H_ * 2);
  unsigned short* xs_bf   = (unsigned short*)take((size_t)TOK * H_ * 2);
  float*          logits  = (float*)take((size_t)TOK * E_ * 4);
  int*            sidx    = (int*)take((size_t)TOK * 4);
  float*          sscore  = (float*)take((size_t)TOK * 4);
  int*            counts  = (int*)take((size_t)E_ * 4);
  int*            offsets = (int*)take((size_t)E_ * 4);
  int*            fillpos = (int*)take((size_t)E_ * 4);
  int*            toklist = (int*)take((size_t)TOK * 4);
  float*          gu_sh   = (float*)take((size_t)TOK * 2 * I_ * 4);
  unsigned short* act_sh  = (unsigned short*)take((size_t)TOK * I_ * 2);
  float*          sh_out  = (float*)take((size_t)TOK * H_ * 4);
  float*          gu_r    = (float*)take((size_t)TOK * 2 * I_ * 4);
  unsigned short* act_r   = (unsigned short*)take((size_t)TOK * I_ * 2);
  float*          routed  = (float*)take((size_t)TOK * H_ * 4);

  dim3 tb(32, 8, 1);
  auto tgrid = [](int N, int K, int Z) { return dim3((N + 31) / 32, (K + 31) / 32, Z); };

  // ---- weight convert/transpose (bf16, [N][K]) ----
  transpose_convert_kernel<<<tgrid(NH_ * D_, H_, 1), tb, 0, stream>>>(wq, wqkv_t, H_, NH_ * D_, 0, 0);
  transpose_convert_kernel<<<tgrid(NKV_ * D_, H_, 1), tb, 0, stream>>>(wk, wqkv_t + (size_t)(NH_ * D_) * H_, H_, NKV_ * D_, 0, 0);
  transpose_convert_kernel<<<tgrid(NKV_ * D_, H_, 1), tb, 0, stream>>>(wv, wqkv_t + (size_t)(NH_ * D_ + NKV_ * D_) * H_, H_, NKV_ * D_, 0, 0);
  transpose_convert_kernel<<<tgrid(H_, H_, 1), tb, 0, stream>>>(wo, wo_t, H_, H_, 0, 0);
  transpose_convert_kernel<<<tgrid(I_, H_, 1), tb, 0, stream>>>(sh_gate, shgu_t, H_, I_, 0, 0);
  transpose_convert_kernel<<<tgrid(I_, H_, 1), tb, 0, stream>>>(sh_up, shgu_t + (size_t)I_ * H_, H_, I_, 0, 0);
  transpose_convert_kernel<<<tgrid(H_, I_, 1), tb, 0, stream>>>(sh_down, shdn_t, I_, H_, 0, 0);
  transpose_convert_kernel<<<tgrid(2 * I_, H_, E_), tb, 0, stream>>>(
      gup, guexp_t, H_, 2 * I_, (size_t)H_ * 2 * I_, (size_t)(2 * I_) * H_);
  transpose_convert_kernel<<<tgrid(H_, I_, E_), tb, 0, stream>>>(
      dnp, dnexp_t, I_, H_, (size_t)I_ * H_, (size_t)H_ * I_);

  // ---- attention half ----
  rmsnorm_kernel<<<TOK, 256, 0, stream>>>(hidden, ln1_w, h1_bf, nullptr);

  gemm_bf16_kernel<<<dim3(QKVN / 128, TOK / 64, 1), 128, 0, stream>>>(
      h1_bf, wqkv_t, qkv, nullptr, TOK, QKVN, H_);

  rope_l2norm_kernel<<<TOK * (NH_ + NKV_), 32, 0, stream>>>(qkv, pos_ids, freqs, q_bf, k_bf);
  v_transpose_kernel<<<dim3(S_ / 256, D_, B_ * NKV_), 256, 0, stream>>>(qkv, vT_bf);

  attn_kernel<<<dim3(S_ / 16, NH_, B_), 32, 0, stream>>>(q_bf, k_bf, vT_bf, attn_bf);

  gemm_bf16_kernel<<<dim3(H_ / 128, TOK / 64, 1), 128, 0, stream>>>(
      attn_bf, wo_t, hid2, hidden, TOK, H_, H_);   // + residual

  // ---- MoE half ----
  rmsnorm_kernel<<<TOK, 256, 0, stream>>>(hid2, ln2_w, h2_bf, h2);

  router_logits_kernel<<<TOK, 256, 0, stream>>>(h2, router_w, logits);
  zero16_kernel<<<1, 32, 0, stream>>>(counts);
  top1_kernel<<<(TOK + 255) / 256, 256, 0, stream>>>(logits, sidx, sscore, counts);
  scan_kernel<<<1, 32, 0, stream>>>(counts, offsets, fillpos);
  fill_kernel<<<(TOK + 255) / 256, 256, 0, stream>>>(sidx, fillpos, toklist);
  scale_tokens_kernel<<<(TOK * H_ + 255) / 256, 256, 0, stream>>>(h2, sscore, xs_bf);

  // shared expert
  gemm_bf16_kernel<<<dim3((2 * I_) / 128, TOK / 64, 1), 128, 0, stream>>>(
      h2_bf, shgu_t, gu_sh, nullptr, TOK, 2 * I_, H_);
  glu_act_kernel<<<(TOK * I_ + 255) / 256, 256, 0, stream>>>(gu_sh, act_sh);
  gemm_bf16_kernel<<<dim3(H_ / 128, TOK / 64, 1), 128, 0, stream>>>(
      act_sh, shdn_t, sh_out, nullptr, TOK, H_, I_);

  // routed experts (gather/scatter, worst-case grid, empty tiles early-out)
  gemm_expert_kernel<<<dim3((2 * I_) / 256, MAXTILES, E_), 128, 0, stream>>>(
      xs_bf, guexp_t, gu_r, toklist, offsets, counts, 2 * I_, H_);
  glu_act_kernel<<<(TOK * I_ + 255) / 256, 256, 0, stream>>>(gu_r, act_r);
  gemm_expert_kernel<<<dim3(H_ / 256, MAXTILES, E_), 128, 0, stream>>>(
      act_r, dnexp_t, routed, toklist, offsets, counts, H_, I_);

  // ---- final residual + shared + routed ----
  final_add_kernel<<<(TOK * H_ + 255) / 256, 256, 0, stream>>>(hid2, sh_out, routed, out);
}
